// RadarBEVFusionWithDCN_79688823210365
// MI455X (gfx1250) — matrix operand order repaired and checked by reading, not compiled, not run on hardware
//
#include <hip/hip_runtime.h>
#include <hip/hip_bf16.h>
#include <math.h>
#include <stdint.h>

typedef __bf16 bf16;
typedef __attribute__((ext_vector_type(16))) __bf16 v16bf;
typedef __attribute__((ext_vector_type(8)))  __bf16 v8bf;
typedef __attribute__((ext_vector_type(8)))  float  v8f;
typedef __attribute__((ext_vector_type(4))) unsigned int u32x4;
typedef __attribute__((ext_vector_type(8))) int i32x8;
typedef __attribute__((ext_vector_type(4))) int i32x4;

#define NB   4
#define HD   128
#define WD   128
#define HW   16384      // 128*128
#define C2   256
#define KTOT 6400       // 25 taps * 256 ch (== 2 groups * 25 taps * 128 ch)

// ---------------------------------------------------------------------------
// WMMA helpers (CDNA5 v_wmma_f32_16x16x32_bf16, wave32)
// A 16x32: lane L row M=L&15; elem e -> K=k0+16*(L>>4)+e.
// B 32x16: lane L col N=L&15; elem e -> K=16*(L>>4)+e (one 32B run per lane).
// C/D:     lane L col N=L&15; elem v -> M=v+8*(L>>4).
// ---------------------------------------------------------------------------
__device__ __forceinline__ v8f wmma_bf16(v16bf a, v16bf b, v8f c) {
  return __builtin_amdgcn_wmma_f32_16x16x32_bf16(false, a, false, b, (short)0, c,
                                                 false, false);
}

// A fragment from row-major bf16 matrix A[M][ldk] (global or LDS pointer).
__device__ __forceinline__ v16bf frag_a(const bf16* A, int ldk, int m0, int k0,
                                        int lane) {
  int hf = lane >> 4;
  const bf16* p = A + (size_t)(m0 + (lane & 15)) * (size_t)ldk + k0 + (hf << 4);
  v8bf lo = *(const v8bf*)p;
  v8bf hi = *(const v8bf*)(p + 8);
  v16bf r;
#pragma unroll
  for (int i = 0; i < 8; ++i) { r[i] = lo[i]; r[i + 8] = hi[i]; }
  return r;
}

// ---------------------------------------------------------------------------
// TDM: async DMA of a 2-D bf16 tile [tile1 rows x tile0 cols] from a row-major
// matrix (leading dim `ld` elements) into LDS at byte offset ldsOff.
// D# per CDNA5 ISA ch.8 (groups 0/1; groups 2/3 zero for 2-D tiles).
// 6-arg builtin form: (g0, g1, g2, g3, g_extra, cpol).
// ---------------------------------------------------------------------------
__device__ __forceinline__ void tdm_load_tile(const bf16* gsrc, unsigned ldsOff,
                                              unsigned tile0, unsigned tile1,
                                              unsigned ld) {
  unsigned long long ga = (unsigned long long)(uintptr_t)gsrc;
  u32x4 g0;
  g0[0] = 1u;                                   // count=1 (valid), user mode
  g0[1] = ldsOff;                               // LDS byte address
  g0[2] = (unsigned)ga;                         // global_addr[31:0]
  g0[3] = (unsigned)((ga >> 32) & 0x01FFFFFFu)  // global_addr[56:32]
          | (2u << 30);                         // type=2 ("image")
  i32x8 g1;
  g1[0] = (int)(1u << 16);                                   // data_size: 2 bytes
  g1[1] = (int)((ld & 0xFFFFu) << 16);                       // tensor_dim0[15:0]
  g1[2] = (int)(((ld >> 16) & 0xFFFFu) |                     // tensor_dim0[31:16]
                ((tile1 & 0xFFFFu) << 16));                  // tensor_dim1[15:0]
  g1[3] = (int)(((tile1 >> 16) & 0xFFFFu) |                  // tensor_dim1[31:16]
                ((tile0 & 0xFFFFu) << 16));                  // tile_dim0
  g1[4] = (int)(tile1 & 0xFFFFu);                            // tile_dim1
  g1[5] = (int)ld;                                           // dim0_stride[31:0]
  g1[6] = 0;
  g1[7] = 0;
  i32x4 gz = {0, 0, 0, 0};
  i32x8 gz8 = {0, 0, 0, 0, 0, 0, 0, 0};
  __builtin_amdgcn_tensor_load_to_lds(g0, g1, gz, gz, gz8, 0);
}

// ---------------------------------------------------------------------------
// Pack concat(bev1,bev2) fp32 NCHW -> bf16 NHWC [B][HW][256]
// ---------------------------------------------------------------------------
__global__ void k_pack(const float* __restrict__ b1, const float* __restrict__ b2,
                       bf16* __restrict__ X, int total) {
  int idx = blockIdx.x * blockDim.x + threadIdx.x;
  if (idx >= total) return;
  int c = idx & 255;
  int rest = idx >> 8;
  int p = rest & (HW - 1);
  int b = rest >> 14;
  float v = (c < 128) ? b1[((size_t)(b * 128 + c)) * HW + p]
                      : b2[((size_t)(b * 128 + (c - 128))) * HW + p];
  X[idx] = (bf16)v;
}

// ---------------------------------------------------------------------------
// Weight prepacks -> row-major bf16 A matrices with NHWC-compatible K order
// ---------------------------------------------------------------------------
__global__ void k_prepack_off(const float* __restrict__ Wsrc, bf16* __restrict__ A,
                              int total) {
  int idx = blockIdx.x * blockDim.x + threadIdx.x;
  if (idx >= total) return;
  int m = idx / KTOT;
  int K = idx - m * KTOT;
  int tap = K >> 8;
  int c = K & 255;
  A[idx] = (m < 100) ? (bf16)Wsrc[((size_t)m * 256 + c) * 25 + tap] : (bf16)0.0f;
}

__global__ void k_prepack_dcn(const float* __restrict__ Wsrc, bf16* __restrict__ A,
                              int total) {
  int idx = blockIdx.x * blockDim.x + threadIdx.x;
  if (idx >= total) return;
  int o = idx / KTOT;
  int K = idx - o * KTOT;
  int g = K / 3200;
  int r = K - g * 3200;
  int tap = r >> 7;
  int cg = r & 127;
  A[idx] = (bf16)Wsrc[((size_t)o * 256 + (g << 7) + cg) * 25 + tap];
}

__global__ void k_prepack_red(const float* __restrict__ Wsrc, bf16* __restrict__ A,
                              int total) {
  int idx = blockIdx.x * blockDim.x + threadIdx.x;
  if (idx >= total) return;
  A[idx] = (bf16)Wsrc[idx];
}

__global__ void k_bnprep(const float* __restrict__ g, const float* __restrict__ bv,
                         const float* __restrict__ m, const float* __restrict__ v,
                         float* __restrict__ sc, float* __restrict__ bi, int n) {
  int i = blockIdx.x * blockDim.x + threadIdx.x;
  if (i >= n) return;
  float inv = g[i] / sqrtf(v[i] + 1e-5f);
  sc[i] = inv;
  bi[i] = bv[i] - m[i] * inv;
}

__global__ void k_zero(float* __restrict__ p, int n) {
  int i = blockIdx.x * blockDim.x + threadIdx.x;
  if (i < n) p[i] = 0.0f;
}

// ---------------------------------------------------------------------------
// Offset-predicting 5x5 conv, implicit GEMM with WMMA.
// A K-chunk tiles (112x32, 7 KB) TDM-staged into double-buffered LDS, shared
// by all 4 waves; per-pixel B fragment register-prefetched one K-step ahead.
// ---------------------------------------------------------------------------
struct OffSmem {
  bf16 aT[2][112 * 32];   // 2 x 7 KB A tiles (TDM destination, offset 0)
};

__global__ __launch_bounds__(128) void k_offset_conv(
    const bf16* __restrict__ X, const bf16* __restrict__ A,
    const float* __restrict__ bias, float* __restrict__ OFF) {
  __shared__ __align__(128) OffSmem sm;

  int tid = threadIdx.x;
  int waveId = tid >> 5;
  int lane = tid & 31;
  int wave = (blockIdx.x << 2) + waveId;
  int b = wave >> 12;
  int p0 = (wave & 4095) << 4;
  int hf = lane >> 4;
  int n = lane & 15;
  int p = p0 + n;
  int y = p >> 7, x = p & 127;

  auto load_b = [&](int kc) -> v16bf {
    int tap = kc >> 3;
    int c0 = (kc & 7) << 5;
    int kh = tap / 5, kw = tap - kh * 5;
    int sy = y + kh - 2, sx = x + kw - 2;
    v16bf r = {};
    if (sy >= 0 && sy < HD && sx >= 0 && sx < WD)
      r = *(const v16bf*)(X + (((size_t)b * HW + sy * WD + sx) << 8) + c0 +
                          (hf << 4));
    return r;
  };

  v8f acc[7];
#pragma unroll
  for (int i = 0; i < 7; ++i) acc[i] = {};

  // ---- prologue: stage chunk 0 ----
  if (waveId == 0) tdm_load_tile(A, 0u, 32u, 112u, KTOT);
  v16bf bcur = load_b(0);
  if (waveId == 0) __builtin_amdgcn_s_wait_tensorcnt(0);
  __syncthreads();

  for (int kc = 0; kc < 200; ++kc) {
    int cur = kc & 1, nxt = cur ^ 1;
    v16bf bnxt = {};
    if (kc + 1 < 200) {
      if (waveId == 0)
        tdm_load_tile(A + (size_t)(kc + 1) * 32,
                      (unsigned)(nxt * (112 * 32 * 2)), 32u, 112u, KTOT);
      bnxt = load_b(kc + 1);         // global load overlaps WMMA burst below
    }
#pragma unroll
    for (int mb = 0; mb < 7; ++mb) {
      v16bf af = frag_a(sm.aT[cur], 32, mb << 4, 0, lane);
      acc[mb] = wmma_bf16(af, bcur, acc[mb]);
    }
    bcur = bnxt;
    if (waveId == 0) __builtin_amdgcn_s_wait_tensorcnt(0);
    __syncthreads();
  }
#pragma unroll
  for (int mb = 0; mb < 7; ++mb) {
#pragma unroll
    for (int v = 0; v < 8; ++v) {
      int row = (mb << 4) + v + (hf << 3);
      if (row < 100)
        OFF[((size_t)b * 100 + row) * HW + p] = acc[mb][v] + bias[row];
    }
  }
}

// ---------------------------------------------------------------------------
// Deformable 5x5 conv main GEMM: out[256 x N] = Aw[256 x 6400] * S[6400 x N]
// N-tile = 32 pixels per workgroup. A K-chunk tiles double-buffered in LDS via
// the Tensor Data Mover (async, wave0-issued); bilinear-sampled B tiles
// double-buffered in LDS. One barrier per K-step. BN+ReLU epilogue.
// ---------------------------------------------------------------------------
struct DcnSmem {
  bf16  aT[2][C2 * 32];     // 2 x 16 KB A tiles (TDM destination, offset 0)
  bf16  bT[2][32 * 32];     // 2 x 2 KB sampled B tiles [n][kk]
  float s_off[100 * 32];    // predicted offsets: 100 ch x 32 px
};

__device__ __forceinline__ void build_btile(const bf16* __restrict__ X,
                                            const float* s_off, bf16* bt, int b,
                                            int p0, int tid, int kc) {
  int g = kc / 100;
  int r = kc - g * 100;
  int tap = r >> 2;                 // 4 chunks of 32 channels per (g,tap)
  int cc = (r & 3) << 5;
  int kh = tap / 5, kw = tap - kh * 5;
  int n = tid & 31;                 // pixel in tile (4 threads per pixel)
  int kkb = (tid >> 5) << 3;        // 8 channel slots per thread
  int bp = p0 + n;
  int by = bp >> 7, bx = bp & 127;
  float py = (float)(by + kh - 2) + s_off[((g * 50) + tap * 2 + 0) * 32 + n];
  float px = (float)(bx + kw - 2) + s_off[((g * 50) + tap * 2 + 1) * 32 + n];
  float fy = floorf(py), fx = floorf(px);
  int y0 = (int)fy, x0 = (int)fx;
  float wy1 = py - fy, wy0 = 1.0f - wy1;
  float wx1 = px - fx, wx0 = 1.0f - wx1;
  int cbase = (g << 7) + cc + kkb;
  float a[8] = {0.f, 0.f, 0.f, 0.f, 0.f, 0.f, 0.f, 0.f};
#pragma unroll
  for (int dy = 0; dy < 2; ++dy) {
#pragma unroll
    for (int dx = 0; dx < 2; ++dx) {
      int yy = y0 + dy, xx = x0 + dx;
      if (yy >= 0 && yy < HD && xx >= 0 && xx < WD) {
        float w = (dy ? wy1 : wy0) * (dx ? wx1 : wx0);
        v8bf v = *(const v8bf*)(X + (((size_t)b * HW + yy * WD + xx) << 8) + cbase);
#pragma unroll
        for (int j = 0; j < 8; ++j) a[j] += w * (float)v[j];
      }
    }
  }
  v8bf o;
#pragma unroll
  for (int j = 0; j < 8; ++j) o[j] = (bf16)a[j];
  *(v8bf*)&bt[(n << 5) + kkb] = o;
}

__global__ __launch_bounds__(128) void k_dcn(
    const bf16* __restrict__ X, const bf16* __restrict__ A,
    const float* __restrict__ OFF, const float* __restrict__ bnscale,
    const float* __restrict__ bnbias, bf16* __restrict__ Y) {
  __shared__ __align__(128) DcnSmem sm;

  int b = blockIdx.x >> 11;          // 2048 n-blocks (of 32 px) per batch
  int nblk = blockIdx.x & 2047;
  int p0 = nblk << 5;
  int tid = threadIdx.x;
  int lane = tid & 31;
  int waveId = tid >> 5;
  int hf = lane >> 4;

  for (int idx = tid; idx < 3200; idx += 128) {   // coalesced 32-float runs
    int ch = idx >> 5, nn = idx & 31;
    sm.s_off[idx] = OFF[((size_t)b * 100 + ch) * HW + p0 + nn];
  }

  v8f acc[4][2];
#pragma unroll
  for (int i = 0; i < 4; ++i) { acc[i][0] = {}; acc[i][1] = {}; }

  __syncthreads();   // s_off ready (needed by B build)

  // ---- prologue: stage chunk 0 (A via TDM, B via sampling) ----
  if (waveId == 0) tdm_load_tile(A, 0u, 32u, (unsigned)C2, KTOT);
  build_btile(X, sm.s_off, sm.bT[0], b, p0, tid, 0);
  if (waveId == 0) __builtin_amdgcn_s_wait_tensorcnt(0);
  __syncthreads();

  for (int kc = 0; kc < 200; ++kc) {
    int cur = kc & 1, nxt = cur ^ 1;
    // stage next chunk (overlaps with WMMA work below)
    if (kc + 1 < 200) {
      if (waveId == 0)
        tdm_load_tile(A + (size_t)(kc + 1) * 32, (unsigned)(nxt * (C2 * 32 * 2)),
                      32u, (unsigned)C2, KTOT);
      build_btile(X, sm.s_off, sm.bT[nxt], b, p0, tid, kc + 1);
    }
    // consume current chunk from LDS
    v16bf b0 = *(const v16bf*)&sm.bT[cur][((lane & 15) << 5) + (hf << 4)];
    v16bf b1 = *(const v16bf*)&sm.bT[cur][((16 + (lane & 15)) << 5) + (hf << 4)];
#pragma unroll
    for (int mb = 0; mb < 4; ++mb) {
      v16bf af = frag_a(sm.aT[cur], 32, (waveId << 6) + (mb << 4), 0, lane);
      acc[mb][0] = wmma_bf16(af, b0, acc[mb][0]);
      acc[mb][1] = wmma_bf16(af, b1, acc[mb][1]);
    }
    if (waveId == 0) __builtin_amdgcn_s_wait_tensorcnt(0);
    __syncthreads();
  }

  // ---- epilogue: BN + ReLU -> bf16 NHWC ----
#pragma unroll
  for (int nb = 0; nb < 2; ++nb) {
    int p = p0 + (nb << 4) + (lane & 15);
#pragma unroll
    for (int mb = 0; mb < 4; ++mb) {
#pragma unroll
      for (int v = 0; v < 8; ++v) {
        int row = (waveId << 6) + (mb << 4) + v + (hf << 3);
        float val = fmaxf(acc[mb][nb][v] * bnscale[row] + bnbias[row], 0.0f);
        Y[(((size_t)b * HW + p) << 8) + row] = (bf16)val;
      }
    }
  }
}

// ---------------------------------------------------------------------------
// SE: channel sums, tiny MLP, channel scale
// ---------------------------------------------------------------------------
__global__ __launch_bounds__(256) void k_chsum(const bf16* __restrict__ Y,
                                               float* __restrict__ sums) {
  int b = blockIdx.x >> 8;
  int pc = blockIdx.x & 255;
  int c = threadIdx.x;
  const bf16* base = Y + (((size_t)b * HW + pc * 64) << 8) + c;
  float s = 0.0f;
  for (int i = 0; i < 64; ++i) s += (float)base[(size_t)i << 8];
  atomicAdd(&sums[(b << 8) + c], s);
}

__global__ __launch_bounds__(256) void k_se_mlp(const float* __restrict__ sums,
                                                const float* __restrict__ w1,
                                                const float* __restrict__ w2,
                                                float* __restrict__ U) {
  __shared__ float t[16];
  int b = blockIdx.x;
  int c = threadIdx.x;
  if (c < 16) {
    float a = 0.0f;
    for (int k = 0; k < 256; ++k)
      a += (sums[(b << 8) + k] * (1.0f / 16384.0f)) * w1[c * 256 + k];
    t[c] = fmaxf(a, 0.0f);
  }
  __syncthreads();
  float a = 0.0f;
#pragma unroll
  for (int j = 0; j < 16; ++j) a += t[j] * w2[c * 16 + j];
  U[(b << 8) + c] = 1.0f / (1.0f + __expf(-a));
}

__global__ void k_se_scale(bf16* __restrict__ Y, const float* __restrict__ U,
                           int total) {
  int idx = blockIdx.x * blockDim.x + threadIdx.x;
  if (idx >= total) return;
  int c = idx & 255;
  int b = idx >> 22;
  Y[idx] = (bf16)((float)Y[idx] * U[(b << 8) + c]);
}

// ---------------------------------------------------------------------------
// Final 1x1 conv (256->128) + BN + ReLU -> fp32 NCHW output.
// Whole A matrix (128x256 bf16 = 64 KB) staged once in LDS via a single TDM
// load; B fragments register-pipelined across the 8 K-chunks.
// ---------------------------------------------------------------------------
__global__ __launch_bounds__(128) void k_reduce(
    const bf16* __restrict__ X, const bf16* __restrict__ A,
    const float* __restrict__ sc, const float* __restrict__ bi,
    float* __restrict__ out) {
  __shared__ __align__(128) bf16 aT[128 * 256];   // 64 KB, single buffer

  int tid = threadIdx.x;
  int waveId = tid >> 5;
  int lane = tid & 31;
  int wave = (blockIdx.x << 2) + waveId;
  int b = wave >> 12;
  int nblk = wave & 4095;
  int p0 = nblk << 4;
  int hf = lane >> 4;
  int n = lane & 15;
  int p = p0 + n;

  if (waveId == 0) {
    tdm_load_tile(A, 0u, 256u, 128u, 256u);   // whole weight matrix -> LDS
    __builtin_amdgcn_s_wait_tensorcnt(0);
  }

  v8f acc[8];
#pragma unroll
  for (int i = 0; i < 8; ++i) acc[i] = {};

  const bf16* xb = X + (((size_t)b * HW + p) << 8) + (hf << 4);
  v16bf bcur = *(const v16bf*)xb;               // chunk 0 B fragment
  __syncthreads();                               // A tile ready

#pragma unroll
  for (int kc = 0; kc < 8; ++kc) {
    v16bf bnxt = {};
    if (kc + 1 < 8) bnxt = *(const v16bf*)(xb + ((kc + 1) << 5));  // prefetch
#pragma unroll
    for (int mb = 0; mb < 8; ++mb) {
      v16bf af = frag_a(aT, 256, mb << 4, kc << 5, lane);
      acc[mb] = wmma_bf16(af, bcur, acc[mb]);
    }
    bcur = bnxt;
  }
#pragma unroll
  for (int mb = 0; mb < 8; ++mb) {
#pragma unroll
    for (int v = 0; v < 8; ++v) {
      int row = (mb << 4) + v + (hf << 3);
      float val = fmaxf(acc[mb][v] * sc[row] + bi[row], 0.0f);
      out[((size_t)(b * 128 + row)) * HW + p] = val;
    }
  }
}

// ---------------------------------------------------------------------------
// Host launcher
// ---------------------------------------------------------------------------
extern "C" void kernel_launch(void* const* d_in, const int* in_sizes, int n_in,
                              void* d_out, int out_size, void* d_ws, size_t ws_size,
                              hipStream_t stream) {
  const float* bev1   = (const float*)d_in[0];
  const float* bev2   = (const float*)d_in[1];
  const float* off_w1 = (const float*)d_in[2];
  const float* off_b1 = (const float*)d_in[3];
  const float* dcn_w1 = (const float*)d_in[4];
  const float* bn1_g  = (const float*)d_in[5];
  const float* bn1_b  = (const float*)d_in[6];
  const float* bn1_m  = (const float*)d_in[7];
  const float* bn1_v  = (const float*)d_in[8];
  const float* se1_w1 = (const float*)d_in[9];
  const float* se1_w2 = (const float*)d_in[10];
  const float* off_w2 = (const float*)d_in[11];
  const float* off_b2 = (const float*)d_in[12];
  const float* dcn_w2 = (const float*)d_in[13];
  const float* bn2_g  = (const float*)d_in[14];
  const float* bn2_b  = (const float*)d_in[15];
  const float* bn2_m  = (const float*)d_in[16];
  const float* bn2_v  = (const float*)d_in[17];
  const float* se2_w1 = (const float*)d_in[18];
  const float* se2_w2 = (const float*)d_in[19];
  const float* red_w  = (const float*)d_in[20];
  const float* bn3_g  = (const float*)d_in[21];
  const float* bn3_b  = (const float*)d_in[22];
  const float* bn3_m  = (const float*)d_in[23];
  const float* bn3_v  = (const float*)d_in[24];
  float* out = (float*)d_out;

  char* ws = (char*)d_ws;
  size_t off = 0;
  auto alloc = [&](size_t bytes) {
    void* p = ws + off;
    off = (off + bytes + 255) & ~(size_t)255;
    return p;
  };
  bf16*  bufA  = (bf16*)alloc((size_t)NB * HW * C2 * 2);
  bf16*  bufB  = (bf16*)alloc((size_t)NB * HW * C2 * 2);
  float* OFFb  = (float*)alloc((size_t)NB * 100 * HW * 4);
  bf16*  Aoff1 = (bf16*)alloc((size_t)112 * KTOT * 2);
  bf16*  Aoff2 = (bf16*)alloc((size_t)112 * KTOT * 2);
  bf16*  Adcn1 = (bf16*)alloc((size_t)256 * KTOT * 2);
  bf16*  Adcn2 = (bf16*)alloc((size_t)256 * KTOT * 2);
  bf16*  Ared  = (bf16*)alloc((size_t)128 * 256 * 2);
  float* bn1s  = (float*)alloc(256 * 4);
  float* bn1bi = (float*)alloc(256 * 4);
  float* bn2s  = (float*)alloc(256 * 4);
  float* bn2bi = (float*)alloc(256 * 4);
  float* bn3s  = (float*)alloc(128 * 4);
  float* bn3bi = (float*)alloc(128 * 4);
  float* MEAN  = (float*)alloc(NB * 256 * 4);
  float* U     = (float*)alloc(NB * 256 * 4);

  const int totalAct = NB * HW * C2;

  k_pack<<<totalAct / 256, 256, 0, stream>>>(bev1, bev2, bufA, totalAct);
  k_bnprep<<<1, 256, 0, stream>>>(bn1_g, bn1_b, bn1_m, bn1_v, bn1s, bn1bi, 256);
  k_bnprep<<<1, 256, 0, stream>>>(bn2_g, bn2_b, bn2_m, bn2_v, bn2s, bn2bi, 256);
  k_bnprep<<<1, 128, 0, stream>>>(bn3_g, bn3_b, bn3_m, bn3_v, bn3s, bn3bi, 128);
  k_prepack_off<<<(112 * KTOT + 255) / 256, 256, 0, stream>>>(off_w1, Aoff1, 112 * KTOT);
  k_prepack_off<<<(112 * KTOT + 255) / 256, 256, 0, stream>>>(off_w2, Aoff2, 112 * KTOT);
  k_prepack_dcn<<<(256 * KTOT + 255) / 256, 256, 0, stream>>>(dcn_w1, Adcn1, 256 * KTOT);
  k_prepack_dcn<<<(256 * KTOT + 255) / 256, 256, 0, stream>>>(dcn_w2, Adcn2, 256 * KTOT);
  k_prepack_red<<<(128 * 256) / 256, 256, 0, stream>>>(red_w, Ared, 128 * 256);

  // ---- layer 1 ----
  k_offset_conv<<<NB * 4096 / 4, 128, 0, stream>>>(bufA, Aoff1, off_b1, OFFb);
  k_dcn<<<NB * 2048, 128, 0, stream>>>(bufA, Adcn1, OFFb, bn1s, bn1bi, bufB);
  k_zero<<<(NB * 256) / 256, 256, 0, stream>>>(MEAN, NB * 256);
  k_chsum<<<NB * 256, 256, 0, stream>>>(bufB, MEAN);
  k_se_mlp<<<NB, 256, 0, stream>>>(MEAN, se1_w1, se1_w2, U);
  k_se_scale<<<totalAct / 256, 256, 0, stream>>>(bufB, U, totalAct);

  // ---- layer 2 ----
  k_offset_conv<<<NB * 4096 / 4, 128, 0, stream>>>(bufB, Aoff2, off_b2, OFFb);
  k_dcn<<<NB * 2048, 128, 0, stream>>>(bufB, Adcn2, OFFb, bn2s, bn2bi, bufA);
  k_zero<<<(NB * 256) / 256, 256, 0, stream>>>(MEAN, NB * 256);
  k_chsum<<<NB * 256, 256, 0, stream>>>(bufA, MEAN);
  k_se_mlp<<<NB, 256, 0, stream>>>(MEAN, se2_w1, se2_w2, U);
  k_se_scale<<<totalAct / 256, 256, 0, stream>>>(bufA, U, totalAct);

  // ---- final 1x1 reduce ----
  k_reduce<<<NB * 4096 / 4, 128, 0, stream>>>(bufA, Ared, bn3s, bn3bi, out);
}
